// SoftPatternClassifier_9861244912258
// MI455X (gfx1250) — compile-verified, matrix operand order checked by reading, not run on hardware
//
#include <hip/hip_runtime.h>
#include <hip/hip_bf16.h>

typedef __attribute__((ext_vector_type(8)))  float  v8f;
typedef __attribute__((ext_vector_type(8)))  __bf16 v8bf;
typedef __attribute__((ext_vector_type(16))) __bf16 v16bf;

#define P_   256
#define L_   6
#define D_   300
#define H_   256
#define C_   5
#define N_   4096
#define NCOL (P_ * 2 * L_)      // 3072
#define KP   320                // K padded to 10 * 32, zero-filled tail
#define NROWS_X 4096
#define NROWS_D 3072
#define NCHUNK 64
#define CHLEN  64               // NCHUNK * CHLEN == N_

__device__ __forceinline__ float sigmoidf(float x) {
    return 1.0f / (1.0f + __expf(-x));
}

__device__ __forceinline__ v16bf cat8(v8bf a, v8bf b) {
    return __builtin_shufflevector(a, b, 0,1,2,3,4,5,6,7,8,9,10,11,12,13,14,15);
}

// -------------------------------------------------------------------------
// Kernel P: gather emb[doc] and diags, split each fp32 value into
// hi/lo bf16 planes (x = hi + lo, lo = bf16(x - float(hi))), K padded
// 300 -> 320 with zeros so all 16B fragment chunks are aligned.
// One block per row (320 threads), rows 0..4095 = X, 4096.. = diags.
// -------------------------------------------------------------------------
__global__ void __launch_bounds__(KP)
split_bf16_kernel(const int* __restrict__ doc,
                  const float* __restrict__ emb,
                  const float* __restrict__ diags,
                  __bf16* __restrict__ Xhi, __bf16* __restrict__ Xlo,
                  __bf16* __restrict__ Dhi, __bf16* __restrict__ Dlo)
{
    const int r = blockIdx.x;
    const int k = threadIdx.x;
    float v = 0.f;
    __bf16* hi; __bf16* lo; long long idx;
    if (r < NROWS_X) {
        if (k < D_) v = emb[(long long)doc[r] * D_ + k];
        idx = (long long)r * KP + k; hi = Xhi; lo = Xlo;
    } else {
        const int rd = r - NROWS_X;
        if (k < D_) v = diags[(long long)rd * D_ + k];
        idx = (long long)rd * KP + k; hi = Dhi; lo = Dlo;
    }
    __bf16 h = (__bf16)v;
    __bf16 l = (__bf16)(v - (float)h);
    hi[idx] = h;
    lo[idx] = l;
}

// -------------------------------------------------------------------------
// Kernel A: bf16-split WMMA GEMM + bias + sigmoid.
//   T[m,n] = sigmoid( X[m,:] . diags[n,:] + bias[n] )
//   X.W ~= Xhi.Whi + Xhi.Wlo + Xlo.Whi   (fp32 accumulate)
// Each wave: one 16-row x 64-col strip (4 tiles), K loop = 10 x 32.
// 16-bit A-frag layout: lane m = l&15; elems 0..7 = K+off(+0..7),
// elems 8..15 = K+off+16, off = (lane<16 ? 0 : 8). B mirrored (col per lane).
// -------------------------------------------------------------------------
__global__ void __launch_bounds__(256)
gemm_sigmoid_kernel(const __bf16* __restrict__ Xhi, const __bf16* __restrict__ Xlo,
                    const __bf16* __restrict__ Dhi, const __bf16* __restrict__ Dlo,
                    const float* __restrict__ bias,
                    float* __restrict__ T)
{
    const int wave = (blockIdx.x * blockDim.x + threadIdx.x) >> 5;
    const int lane = threadIdx.x & 31;
    const int mt = wave / (NCOL / 64);      // 0..255  row tile (16 rows)
    const int ng = wave % (NCOL / 64);      // 0..47   group of 4 col tiles
    const int nlane = lane & 15;
    const int off = (lane < 16) ? 0 : 8;

    const long long arow = (long long)(mt * 16 + nlane) * KP;
    const __bf16* __restrict__ ah = Xhi + arow;
    const __bf16* __restrict__ al = Xlo + arow;

    const long long b0r = (long long)(ng * 64 +  0 + nlane) * KP;
    const long long b1r = (long long)(ng * 64 + 16 + nlane) * KP;
    const long long b2r = (long long)(ng * 64 + 32 + nlane) * KP;
    const long long b3r = (long long)(ng * 64 + 48 + nlane) * KP;

    v8f acc0 = {}, acc1 = {}, acc2 = {}, acc3 = {};

    #pragma unroll 2
    for (int k = 0; k < KP; k += 32) {
        const int kb = k + off;
        v16bf Ah = cat8(*(const v8bf*)(ah + kb), *(const v8bf*)(ah + kb + 16));
        v16bf Al = cat8(*(const v8bf*)(al + kb), *(const v8bf*)(al + kb + 16));

        v16bf Bh0 = cat8(*(const v8bf*)(Dhi + b0r + kb), *(const v8bf*)(Dhi + b0r + kb + 16));
        v16bf Bl0 = cat8(*(const v8bf*)(Dlo + b0r + kb), *(const v8bf*)(Dlo + b0r + kb + 16));
        v16bf Bh1 = cat8(*(const v8bf*)(Dhi + b1r + kb), *(const v8bf*)(Dhi + b1r + kb + 16));
        v16bf Bl1 = cat8(*(const v8bf*)(Dlo + b1r + kb), *(const v8bf*)(Dlo + b1r + kb + 16));
        v16bf Bh2 = cat8(*(const v8bf*)(Dhi + b2r + kb), *(const v8bf*)(Dhi + b2r + kb + 16));
        v16bf Bl2 = cat8(*(const v8bf*)(Dlo + b2r + kb), *(const v8bf*)(Dlo + b2r + kb + 16));
        v16bf Bh3 = cat8(*(const v8bf*)(Dhi + b3r + kb), *(const v8bf*)(Dhi + b3r + kb + 16));
        v16bf Bl3 = cat8(*(const v8bf*)(Dlo + b3r + kb), *(const v8bf*)(Dlo + b3r + kb + 16));

        acc0 = __builtin_amdgcn_wmma_f32_16x16x32_bf16(false, Ah, false, Bh0, (short)0, acc0, false, false);
        acc1 = __builtin_amdgcn_wmma_f32_16x16x32_bf16(false, Ah, false, Bh1, (short)0, acc1, false, false);
        acc2 = __builtin_amdgcn_wmma_f32_16x16x32_bf16(false, Ah, false, Bh2, (short)0, acc2, false, false);
        acc3 = __builtin_amdgcn_wmma_f32_16x16x32_bf16(false, Ah, false, Bh3, (short)0, acc3, false, false);
        acc0 = __builtin_amdgcn_wmma_f32_16x16x32_bf16(false, Ah, false, Bl0, (short)0, acc0, false, false);
        acc1 = __builtin_amdgcn_wmma_f32_16x16x32_bf16(false, Ah, false, Bl1, (short)0, acc1, false, false);
        acc2 = __builtin_amdgcn_wmma_f32_16x16x32_bf16(false, Ah, false, Bl2, (short)0, acc2, false, false);
        acc3 = __builtin_amdgcn_wmma_f32_16x16x32_bf16(false, Ah, false, Bl3, (short)0, acc3, false, false);
        acc0 = __builtin_amdgcn_wmma_f32_16x16x32_bf16(false, Al, false, Bh0, (short)0, acc0, false, false);
        acc1 = __builtin_amdgcn_wmma_f32_16x16x32_bf16(false, Al, false, Bh1, (short)0, acc1, false, false);
        acc2 = __builtin_amdgcn_wmma_f32_16x16x32_bf16(false, Al, false, Bh2, (short)0, acc2, false, false);
        acc3 = __builtin_amdgcn_wmma_f32_16x16x32_bf16(false, Al, false, Bh3, (short)0, acc3, false, false);
    }

    // Epilogue: C/D layout -> lane l, vgpr v holds (M = v + (l<16?0:8), N = l&15)
    const int rbase = mt * 16 + ((lane < 16) ? 0 : 8);
    #pragma unroll
    for (int j = 0; j < 4; ++j) {
        const v8f acc = (j == 0) ? acc0 : (j == 1) ? acc1 : (j == 2) ? acc2 : acc3;
        const int ncol = ng * 64 + j * 16 + nlane;
        const float bs = bias[ncol];
        #pragma unroll
        for (int v = 0; v < 8; ++v) {
            float x = acc[v] + bs;
            T[(long long)(rbase + v) * NCOL + ncol] = sigmoidf(x);
        }
    }
}

// -------------------------------------------------------------------------
// Kernel B: chunked affine scan. For each (pattern p, chunk c) run the
// 64-step recurrence on 7 basis inputs:
//   j = 0..5 : h_in = e_j, no +1 constant  -> column j of chunk matrix A
//   j = 6    : h_in = 0, with +1 constant  -> offset c and score constant e
// Output coef[c][p][j] = { h_final[0..5], s } (7 floats).
// -------------------------------------------------------------------------
__global__ void __launch_bounds__(256)
scan_chunks_kernel(const float* __restrict__ T,
                   const float* __restrict__ epsilon,
                   float* __restrict__ coef)
{
    const int g = blockIdx.x * blockDim.x + threadIdx.x;
    if (g >= P_ * NCHUNK * 7) return;
    const int j  = g % 7;
    const int pc = g / 7;
    const int p  = pc & (P_ - 1);
    const int c  = pc >> 8;

    const float sls = 0.622459331201855f;   // sigmoid(0.5)
    float ev[L_ - 1];
    #pragma unroll
    for (int l = 0; l < L_ - 1; ++l)
        ev[l] = sls * sigmoidf(epsilon[p * (L_ - 1) + l]);

    float h[L_] = {0.f, 0.f, 0.f, 0.f, 0.f, 0.f};
    float addc = 0.f;
    if (j < L_) h[j] = 1.f; else addc = 1.f;
    float s = 0.f;

    const float* __restrict__ Tp = T + (long long)(c * CHLEN) * NCOL + p * (2 * L_);
    for (int i = 0; i < CHLEN; ++i) {
        const float* tr = Tp + (long long)i * NCOL;
        // 12 contiguous floats, base address multiple of 48B -> 16B aligned
        const float4 q0 = *(const float4*)(tr);
        const float4 q1 = *(const float4*)(tr + 4);
        const float4 q2 = *(const float4*)(tr + 8);
        const float T0[L_] = {q0.x, q0.y, q0.z, q0.w, q1.x, q1.y};
        const float T1[L_] = {q1.z, q1.w, q2.x, q2.y, q2.z, q2.w};

        float mn[L_];
        mn[0] = sls * h[0] * T0[0];
        #pragma unroll
        for (int q = 1; q < L_; ++q)
            mn[q] = h[q - 1] * T1[q - 1] + sls * h[q] * T0[q];

        float h2[L_];
        h2[0] = mn[0] + addc;
        #pragma unroll
        for (int q = 1; q < L_; ++q)
            h2[q] = mn[q] + mn[q - 1] * ev[q - 1];

        s += h2[L_ - 1];
        #pragma unroll
        for (int l = 0; l < L_; ++l) h[l] = h2[l];
    }

    float* out = coef + ((long long)(c * P_ + p) * 7 + j) * 7;
    #pragma unroll
    for (int l = 0; l < L_; ++l) out[l] = h[l];
    out[L_] = s;
}

// -------------------------------------------------------------------------
// Kernel C: sequentially compose the 64 chunk affine maps per pattern.
//   h_out = A h + c ;  s += d.h + e      -> scores[p]
// -------------------------------------------------------------------------
__global__ void __launch_bounds__(256)
combine_kernel(const float* __restrict__ coef, float* __restrict__ scores)
{
    const int p = threadIdx.x;
    float h[L_] = {1.f, 0.f, 0.f, 0.f, 0.f, 0.f};   // h0[:,0] = 1
    float s = 0.f;
    for (int c = 0; c < NCHUNK; ++c) {
        const float* base = coef + (long long)(c * P_ + p) * 7 * 7;
        const float* aff  = base + 6 * 7;
        float nh[L_];
        #pragma unroll
        for (int i = 0; i < L_; ++i) nh[i] = aff[i];
        float ds = aff[L_];
        #pragma unroll
        for (int k = 0; k < L_; ++k) {
            const float hk = h[k];
            const float* col = base + k * 7;
            #pragma unroll
            for (int i = 0; i < L_; ++i) nh[i] += col[i] * hk;
            ds += col[L_] * hk;
        }
        #pragma unroll
        for (int i = 0; i < L_; ++i) h[i] = nh[i];
        s += ds;
    }
    scores[p] = s;
}

// -------------------------------------------------------------------------
// Kernel D: 3-layer MLP head on scores (1x256 -> 256 -> 256 -> 5).
// -------------------------------------------------------------------------
__global__ void __launch_bounds__(256)
mlp_kernel(const float* __restrict__ scores,
           const float* __restrict__ w0, const float* __restrict__ b0,
           const float* __restrict__ w1, const float* __restrict__ b1,
           const float* __restrict__ w2, const float* __restrict__ b2,
           float* __restrict__ out)
{
    __shared__ float sc[H_], z0[H_], z1[H_];
    const int t = threadIdx.x;
    sc[t] = scores[t];
    __syncthreads();

    float acc = b0[t];
    for (int k = 0; k < P_; ++k) acc = fmaf(w0[t * P_ + k], sc[k], acc);
    z0[t] = fmaxf(acc, 0.f);
    __syncthreads();

    acc = b1[t];
    for (int k = 0; k < H_; ++k) acc = fmaf(w1[t * H_ + k], z0[k], acc);
    z1[t] = fmaxf(acc, 0.f);
    __syncthreads();

    if (t < C_) {
        float a = b2[t];
        for (int k = 0; k < H_; ++k) a = fmaf(w2[t * H_ + k], z1[k], a);
        out[t] = a;
    }
}

extern "C" void kernel_launch(void* const* d_in, const int* in_sizes, int n_in,
                              void* d_out, int out_size, void* d_ws, size_t ws_size,
                              hipStream_t stream)
{
    const int*   doc     = (const int*)  d_in[0];
    const float* emb     = (const float*)d_in[1];
    const float* diags   = (const float*)d_in[2];
    const float* bias    = (const float*)d_in[3];
    const float* epsilon = (const float*)d_in[4];
    const float* w0      = (const float*)d_in[5];
    const float* b0      = (const float*)d_in[6];
    const float* w1      = (const float*)d_in[7];
    const float* b1      = (const float*)d_in[8];
    const float* w2      = (const float*)d_in[9];
    const float* b2      = (const float*)d_in[10];
    float* out = (float*)d_out;

    // workspace layout
    char* ws = (char*)d_ws;
    size_t off = 0;
    float* T      = (float*)(ws + off); off += (size_t)N_ * NCOL * sizeof(float);     // 48 MB
    float* coef   = (float*)(ws + off); off += (size_t)NCHUNK * P_ * 49 * sizeof(float);
    float* scores = (float*)(ws + off); off += (size_t)P_ * sizeof(float);
    off = (off + 255) & ~(size_t)255;
    __bf16* Xhi = (__bf16*)(ws + off); off += (size_t)NROWS_X * KP * sizeof(__bf16);
    __bf16* Xlo = (__bf16*)(ws + off); off += (size_t)NROWS_X * KP * sizeof(__bf16);
    __bf16* Dhi = (__bf16*)(ws + off); off += (size_t)NROWS_D * KP * sizeof(__bf16);
    __bf16* Dlo = (__bf16*)(ws + off); off += (size_t)NROWS_D * KP * sizeof(__bf16);

    // P: gather + hi/lo bf16 split (rows: 4096 X + 3072 diags)
    split_bf16_kernel<<<NROWS_X + NROWS_D, KP, 0, stream>>>(doc, emb, diags, Xhi, Xlo, Dhi, Dlo);

    // A: 256 mtiles * 48 ntile-groups = 12288 waves, 8 waves/block -> 1536 blocks
    gemm_sigmoid_kernel<<<1536, 256, 0, stream>>>(Xhi, Xlo, Dhi, Dlo, bias, T);

    // B: 256 * 64 * 7 = 114688 threads
    scan_chunks_kernel<<<(P_ * NCHUNK * 7) / 256, 256, 0, stream>>>(T, epsilon, coef);

    // C: one block, one thread per pattern
    combine_kernel<<<1, 256, 0, stream>>>(coef, scores);

    // D: MLP head
    mlp_kernel<<<1, 256, 0, stream>>>(scores, w0, b0, w1, b1, w2, b2, out);
}
